// FullAttention_42872363549271
// MI455X (gfx1250) — matrix operand order repaired and checked by reading, not compiled, not run on hardware
//
#include <hip/hip_runtime.h>
#include <hip/hip_bf16.h>

// ---------------------------------------------------------------------------
// Full multi-head attention (fwd) for MI455X / gfx1250, wave32 + WMMA bf16.
// Staging: double-buffered direct memory->LDS global_load_async_to_lds_b128
// (ASYNCcnt) software pipeline overlapping DMA of tile t+1 with WMMA on t.
//   out  = (softmax((xWq^T+bq)(xWk^T+bk)^T / 8) (xWv^T+bv)) Wo^T + bo
// d_out = [ out : 2*2048*1024 f32 | attn_weights : 2*16*2048*2048 f32 ]
// ---------------------------------------------------------------------------

typedef __bf16 bf16;
typedef __attribute__((ext_vector_type(16))) __bf16 v16bf;
typedef __attribute__((ext_vector_type(8)))  __bf16 v8bf;
typedef __attribute__((ext_vector_type(8)))  float  v8f;

#define BATCH   2
#define N_CTX   2048
#define D_MODEL 1024
#define N_HEADS 16
#define D_HEAD  64
#define N_ROWS  (BATCH * N_CTX)          // 4096  (flattened token rows)
#define N_BH    (BATCH * N_HEADS)        // 32    (batched-head count)

enum { EPI_QK = 0, EPI_VT = 1, EPI_ATTN = 2, EPI_CTX = 3, EPI_OUT = 4 };

__device__ __forceinline__ bf16 f2bf(float f) {
    union { float f; unsigned u; } x; x.f = f;
    unsigned r = x.u + 0x7FFFu + ((x.u >> 16) & 1u);   // round-to-nearest-even
    union { unsigned short s; bf16 b; } y;
    y.s = (unsigned short)(r >> 16);
    return y.b;
}

// Low 32 bits of a generic shared-space pointer == LDS byte offset.
__device__ __forceinline__ unsigned lds_off(const void* p) {
    return (unsigned)(unsigned long long)p;
}

// Direct global->LDS async copy, 16B per lane; tracked by ASYNCcnt.
__device__ __forceinline__ void async_copy_b128(void* lds_dst, const void* gsrc) {
    const unsigned           l = lds_off(lds_dst);
    const unsigned long long g = (unsigned long long)gsrc;
    asm volatile("global_load_async_to_lds_b128 %0, %1, off"
                 :: "v"(l), "v"(g) : "memory");
}

__device__ __forceinline__ void wait_async_all() {
    asm volatile("s_wait_asynccnt 0x0" ::: "memory");
}

// ---------------------------------------------------------------------------
// fp32 -> bf16 conversion (grid-stride)
// ---------------------------------------------------------------------------
__global__ void cvt_f32_bf16(const float* __restrict__ s, bf16* __restrict__ d, int n) {
    for (int i = blockIdx.x * blockDim.x + threadIdx.x; i < n;
         i += gridDim.x * blockDim.x)
        d[i] = f2bf(s[i]);
}

// ---------------------------------------------------------------------------
// Generic WMMA GEMM:  C[M,N] = A[M,K] * Bmat[N,K]^T   (+ epilogue)
//   256 threads = 8 wave32, wave grid WR x WC, per-wave MT x NT 16x16 tiles.
//   BK == 32 == one v_wmma_f32_16x16x32_bf16 K-step.
//   Double-buffered LDS; bf16 operands DMA'd via async b128 copies, fp32 A
//   (PV pass) converts through VGPRs with float4 loads + packed b128 store.
// ---------------------------------------------------------------------------
template <int BM, int BN, int BK, int WR, int WC, int MT, int NT, bool AF32, int EPI>
__global__ __launch_bounds__(256) void gemm_wmma_kernel(
    const void* __restrict__ Av, const bf16* __restrict__ Bmat,
    const float* __restrict__ bias, void* __restrict__ Cv,
    int K, int lda, int ldb,
    long long aBatch, long long bBatch, long long cBatch)
{
    static_assert(WR * WC == 8, "8 waves per block");
    static_assert(WR * MT * 16 == BM && WC * NT * 16 == BN, "tiling");
    constexpr int LDA = BK + 8;   // 80B rows: 16B-aligned, bank-staggered
    constexpr int LDB = BK + 8;
    constexpr int CPR = BK / 8;            // 8-elem chunks per row
    constexpr int ACH = BM * BK / 8;       // total A chunks (multiple of 256)
    constexpr int BCH = BN * BK / 8;       // total B chunks (multiple of 256)
    static_assert(ACH % 256 == 0 && BCH % 256 == 0, "staging divisibility");

    __shared__ __align__(16) bf16 Alds[2][BM * LDA];
    __shared__ __align__(16) bf16 Blds[2][BN * LDB];

    const int tid  = threadIdx.x;
    const int lane = tid & 31;
    const int wave = tid >> 5;
    const int wr   = wave / WC;
    const int wc   = wave % WC;
    const int z    = blockIdx.z;
    const int rowBase = blockIdx.y * BM;
    const int colBase = blockIdx.x * BN;

    const float* Af = (const float*)Av + (size_t)z * (size_t)aBatch;
    const bf16*  Ab = (const bf16*)Av  + (size_t)z * (size_t)aBatch;
    const bf16*  Bp = Bmat             + (size_t)z * (size_t)bBatch;

    // Stage one BKxBM / BKxBN tile pair into LDS buffer `buf` at K-offset kt.
    auto stage = [&](int buf, int kt) {
#pragma unroll
        for (int it = 0; it < ACH / 256; ++it) {
            const int c  = tid + it * 256;
            const int r  = c / CPR;
            const int cc = (c % CPR) * 8;
            if constexpr (AF32) {
                const float4* s4 = (const float4*)(Af + (size_t)(rowBase + r) * (size_t)lda + kt + cc);
                const float4 u = s4[0];
                const float4 w = s4[1];
                v8bf pk;
                pk[0] = f2bf(u.x); pk[1] = f2bf(u.y); pk[2] = f2bf(u.z); pk[3] = f2bf(u.w);
                pk[4] = f2bf(w.x); pk[5] = f2bf(w.y); pk[6] = f2bf(w.z); pk[7] = f2bf(w.w);
                *(v8bf*)&Alds[buf][r * LDA + cc] = pk;
            } else {
                async_copy_b128(&Alds[buf][r * LDA + cc],
                                Ab + (size_t)(rowBase + r) * (size_t)lda + kt + cc);
            }
        }
#pragma unroll
        for (int it = 0; it < BCH / 256; ++it) {
            const int c  = tid + it * 256;
            const int r  = c / CPR;
            const int cc = (c % CPR) * 8;
            async_copy_b128(&Blds[buf][r * LDB + cc],
                            Bp + (size_t)(colBase + r) * (size_t)ldb + kt + cc);
        }
    };

    v8f zero;
#pragma unroll
    for (int e = 0; e < 8; ++e) zero[e] = 0.0f;

    v8f acc[MT][NT];
#pragma unroll
    for (int i = 0; i < MT; ++i)
#pragma unroll
        for (int j = 0; j < NT; ++j) acc[i][j] = zero;

    const int klo  = (lane >> 4) * 8;  // A-frag K half select (ISA 7.12.2)
    const int rsel = lane & 15;

    const int nk = K / BK;
    stage(0, 0);                       // prologue: fill buffer 0

    for (int ki = 0; ki < nk; ++ki) {
        const int cur = ki & 1;
        wait_async_all();              // copies issued for buf `cur` landed
        __syncthreads();               // all waves staged / done reading cur^1

        if (ki + 1 < nk) stage(cur ^ 1, (ki + 1) * BK);   // overlap next DMA

        // ---- load fragments from buffer `cur` ----
        v16bf afrag[MT], bfrag[NT];
#pragma unroll
        for (int i = 0; i < MT; ++i) {
            const bf16* ar = &Alds[cur][((wr * MT + i) * 16 + rsel) * LDA];
#pragma unroll
            for (int j = 0; j < 8; ++j) {
                afrag[i][j]     = ar[klo + j];        // K in {0..7} or {8..15}
                afrag[i][8 + j] = ar[16 + klo + j];   // K in {16..23} or {24..31}
            }
        }
#pragma unroll
        for (int j = 0; j < NT; ++j) {
            const bf16* br = &Blds[cur][((wc * NT + j) * 16 + rsel) * LDB + ((lane >> 4) * 8)];
#pragma unroll
            for (int e = 0; e < 16; ++e) bfrag[j][e] = br[e + (e & 8)];
        }

        // ---- matrix ops ----
#pragma unroll
        for (int i = 0; i < MT; ++i)
#pragma unroll
            for (int j = 0; j < NT; ++j)
                acc[i][j] = __builtin_amdgcn_wmma_f32_16x16x32_bf16(
                    false, afrag[i], false, bfrag[j],
                    (short)0, acc[i][j], false, false);
    }

    // ---- epilogue (C/D layout: VGPR r -> M = r + 8*(lane>=16), N = lane&15) --
    const int mrow0 = (lane >> 4) * 8;
    const int ncol  = lane & 15;
#pragma unroll
    for (int i = 0; i < MT; ++i) {
#pragma unroll
        for (int j = 0; j < NT; ++j) {
            const int n = colBase + (wc * NT + j) * 16 + ncol;
#pragma unroll
            for (int r = 0; r < 8; ++r) {
                const int m = rowBase + (wr * MT + i) * 16 + mrow0 + r;
                const float c = acc[i][j][r];
                if constexpr (EPI == EPI_QK) {
                    // Q/K: bf16 scatter to [B,H,n_ctx,d_head], + bias
                    const int b = m >> 11, nq = m & (N_CTX - 1);
                    const int h = n >> 6,  d  = n & (D_HEAD - 1);
                    ((bf16*)Cv)[(((size_t)(b * N_HEADS + h) * N_CTX + nq) * D_HEAD) + d] =
                        f2bf(c + bias[n]);
                } else if constexpr (EPI == EPI_VT) {
                    // V: bf16 scatter transposed to [B,H,d_head,n_ctx], + bias
                    const int b = m >> 11, nq = m & (N_CTX - 1);
                    const int h = n >> 6,  d  = n & (D_HEAD - 1);
                    ((bf16*)Cv)[(((size_t)(b * N_HEADS + h) * D_HEAD + d) * N_CTX) + nq] =
                        f2bf(c + bias[n]);
                } else if constexpr (EPI == EPI_ATTN) {
                    // raw scores * 1/sqrt(64) -> fp32 attn area of d_out
                    ((float*)Cv)[(size_t)z * (size_t)cBatch + (size_t)m * N_CTX + n] =
                        c * 0.125f;
                } else if constexpr (EPI == EPI_CTX) {
                    // context bf16 -> [B, n_ctx, D_MODEL] (col = h*64 + d)
                    const int b = z >> 4, h = z & (N_HEADS - 1);
                    ((bf16*)Cv)[((size_t)(b * N_CTX + m)) * D_MODEL + h * D_HEAD + n] =
                        f2bf(c);
                } else {
                    // final out: fp32 + bias
                    ((float*)Cv)[(size_t)m * D_MODEL + n] = c + bias[n];
                }
            }
        }
    }
}

// ---------------------------------------------------------------------------
// Row softmax over 2048-wide rows, one 256-thread block per row.
// ---------------------------------------------------------------------------
__global__ __launch_bounds__(256) void softmax_rows(float* __restrict__ attn) {
    __shared__ float red[8];
    float* p = attn + (size_t)blockIdx.x * N_CTX;
    const int t = threadIdx.x;

    float v[8];
    float m = -3.4e38f;
#pragma unroll
    for (int i = 0; i < 8; ++i) { v[i] = p[t + i * 256]; m = fmaxf(m, v[i]); }
#pragma unroll
    for (int o = 16; o > 0; o >>= 1) m = fmaxf(m, __shfl_xor(m, o, 32));
    if ((t & 31) == 0) red[t >> 5] = m;
    __syncthreads();
    float bm = red[0];
#pragma unroll
    for (int w = 1; w < 8; ++w) bm = fmaxf(bm, red[w]);

    float s = 0.0f;
#pragma unroll
    for (int i = 0; i < 8; ++i) { v[i] = __expf(v[i] - bm); s += v[i]; }
#pragma unroll
    for (int o = 16; o > 0; o >>= 1) s += __shfl_xor(s, o, 32);
    __syncthreads();
    if ((t & 31) == 0) red[t >> 5] = s;
    __syncthreads();
    float bs = 0.0f;
#pragma unroll
    for (int w = 0; w < 8; ++w) bs += red[w];
    const float inv = 1.0f / bs;
#pragma unroll
    for (int i = 0; i < 8; ++i) p[t + i * 256] = v[i] * inv;
}

// ---------------------------------------------------------------------------
// Launch
// ---------------------------------------------------------------------------
extern "C" void kernel_launch(void* const* d_in, const int* in_sizes, int n_in,
                              void* d_out, int out_size, void* d_ws, size_t ws_size,
                              hipStream_t stream) {
    const float* x  = (const float*)d_in[0];
    const float* Wq = (const float*)d_in[1];
    const float* bq = (const float*)d_in[2];
    const float* Wk = (const float*)d_in[3];
    const float* bk = (const float*)d_in[4];
    const float* Wv = (const float*)d_in[5];
    const float* bv = (const float*)d_in[6];
    const float* Wo = (const float*)d_in[7];
    const float* bo = (const float*)d_in[8];

    float* outp = (float*)d_out;
    float* attn = outp + (size_t)N_ROWS * D_MODEL;          // attn_weights region

    // workspace layout (bf16 elements): 48 MiB total
    bf16* ws = (bf16*)d_ws;
    size_t o = 0;
    bf16* xbf  = ws + o; o += (size_t)N_ROWS * D_MODEL;       // 4096x1024
    bf16* wqb  = ws + o; o += (size_t)D_MODEL * D_MODEL;
    bf16* wkb  = ws + o; o += (size_t)D_MODEL * D_MODEL;
    bf16* wvb  = ws + o; o += (size_t)D_MODEL * D_MODEL;
    bf16* wob  = ws + o; o += (size_t)D_MODEL * D_MODEL;
    bf16* qbf  = ws + o; o += (size_t)N_BH * N_CTX * D_HEAD;  // [B,H,n,64]
    bf16* kbf  = ws + o; o += (size_t)N_BH * N_CTX * D_HEAD;  // [B,H,n,64]
    bf16* vtbf = ws + o; o += (size_t)N_BH * D_HEAD * N_CTX;  // [B,H,64,n]
    bf16* ctxb = ws + o; o += (size_t)N_ROWS * D_MODEL;       // [B,n,1024]

    const dim3 blk(256);

    // 1) fp32 -> bf16 conversions
    cvt_f32_bf16<<<2048, 256, 0, stream>>>(x,  xbf, N_ROWS * D_MODEL);
    cvt_f32_bf16<<<1024, 256, 0, stream>>>(Wq, wqb, D_MODEL * D_MODEL);
    cvt_f32_bf16<<<1024, 256, 0, stream>>>(Wk, wkb, D_MODEL * D_MODEL);
    cvt_f32_bf16<<<1024, 256, 0, stream>>>(Wv, wvb, D_MODEL * D_MODEL);
    cvt_f32_bf16<<<1024, 256, 0, stream>>>(Wo, wob, D_MODEL * D_MODEL);

    // 2) projections: [4096,1024] = xbf @ W^T + b   (grid: N/128 x M/128)
    gemm_wmma_kernel<128,128,32, 2,4, 4,2, false, EPI_QK>
        <<<dim3(8, 32, 1), blk, 0, stream>>>(xbf, wqb, bq, qbf,
            D_MODEL, D_MODEL, D_MODEL, 0, 0, 0);
    gemm_wmma_kernel<128,128,32, 2,4, 4,2, false, EPI_QK>
        <<<dim3(8, 32, 1), blk, 0, stream>>>(xbf, wkb, bk, kbf,
            D_MODEL, D_MODEL, D_MODEL, 0, 0, 0);
    gemm_wmma_kernel<128,128,32, 2,4, 4,2, false, EPI_VT>
        <<<dim3(8, 32, 1), blk, 0, stream>>>(xbf, wvb, bv, vtbf,
            D_MODEL, D_MODEL, D_MODEL, 0, 0, 0);

    // 3) scores = Q K^T / 8 : batched over z = b*H+h, K-dim = 64
    gemm_wmma_kernel<128,128,32, 2,4, 4,2, false, EPI_ATTN>
        <<<dim3(N_CTX/128, N_CTX/128, N_BH), blk, 0, stream>>>(qbf, kbf, nullptr, attn,
            D_HEAD, D_HEAD, D_HEAD,
            (long long)N_CTX * D_HEAD, (long long)N_CTX * D_HEAD,
            (long long)N_CTX * N_CTX);

    // 4) softmax over each 2048-length row (B*H*n rows)
    softmax_rows<<<N_BH * N_CTX, 256, 0, stream>>>(attn);

    // 5) context = P @ V : batched, A = attn fp32 (converted in staging), B = V^T
    gemm_wmma_kernel<128,64,32, 4,2, 2,2, true, EPI_CTX>
        <<<dim3(1, N_CTX/128, N_BH), blk, 0, stream>>>(attn, vtbf, nullptr, ctxb,
            N_CTX, N_CTX, N_CTX,
            (long long)N_CTX * N_CTX, (long long)D_HEAD * N_CTX, 0);

    // 6) out = context @ Wo^T + bo  -> fp32 d_out
    gemm_wmma_kernel<128,128,32, 2,4, 4,2, false, EPI_OUT>
        <<<dim3(8, 32, 1), blk, 0, stream>>>(ctxb, wob, bo, outp,
            D_MODEL, D_MODEL, D_MODEL, 0, 0, 0);
}
